// SelfAttentionWithShift_46385646797250
// MI455X (gfx1250) — compile-verified, matrix operand order checked by reading, not run on hardware
//
#include <hip/hip_runtime.h>

#define HIDDEN   1024
#define HEADS    16
#define HEAD_DIM 64
#define BATCH    2
#define SEQ      1024
#define M_TOT    (BATCH * SEQ)
#define SPAD     (SEQ + 8)   // padded LDS row stride (floats) to break bank conflicts

typedef __bf16 bf16_t;
typedef __attribute__((ext_vector_type(16))) __bf16        v16bf;
typedef __attribute__((ext_vector_type(8)))  float         v8f;
typedef __attribute__((ext_vector_type(4)))  unsigned int  u32x4;
typedef __attribute__((ext_vector_type(8)))  int           i32x8;
typedef __attribute__((ext_vector_type(4)))  int           i32x4;

#if defined(__has_builtin)
#if __has_builtin(__builtin_amdgcn_tensor_load_to_lds)
#define HAVE_TDM 1
#endif
#endif

// ---------- helpers ----------
static __device__ __forceinline__ bf16_t f2bf(float x) {
  unsigned u = __builtin_bit_cast(unsigned, x);
  unsigned r = (u + 0x7FFFu + ((u >> 16) & 1u)) >> 16;  // round-to-nearest-even
  unsigned short s = (unsigned short)r;
  return __builtin_bit_cast(bf16_t, s);
}

// 16x32 bf16 A/B fragment from a row-major bf16 matrix.
// Lane L: row = row0 + (L&15); K chunks {k0+off..+7} and {k0+off+16..+23}, off = 8*(L>=16).
// Two contiguous 16B chunks per lane -> 2x global_load_b128 (or ds_load_b128 from LDS).
static __device__ __forceinline__ v16bf frag_from_bf16(const bf16_t* __restrict__ base,
                                                       int row0, int stride, int k0, int lane) {
  const bf16_t* p = base + (size_t)(row0 + (lane & 15)) * stride + k0 + ((lane >> 4) << 3);
  v16bf f;
#pragma unroll
  for (int i = 0; i < 8; ++i) { f[i] = p[i]; f[i + 8] = p[i + 16]; }
  return f;
}

// ---------- fp32 -> bf16 conversion pass (amortizes all conversion VALU) ----------
__global__ __launch_bounds__(256) void cvt_kernel(const float* __restrict__ src,
                                                  bf16_t* __restrict__ dst, int n) {
  int i = (blockIdx.x * 256 + threadIdx.x) * 4;
  if (i + 3 < n) {
    float4 f = *(const float4*)(src + i);
    dst[i + 0] = f2bf(f.x);
    dst[i + 1] = f2bf(f.y);
    dst[i + 2] = f2bf(f.z);
    dst[i + 3] = f2bf(f.w);
  }
}

// ---------- GEMM: out = X(M x K) @ W^T + bias; X,W bf16; W is (N,K) row-major ----------
// Each wave computes a 16x64 strip: one A fragment feeds 4 WMMAs per K-step.
// mode 0: dst bf16, head layout      qh[(b*H+h)*S*Dh + s*Dh + d]
// mode 1: dst bf16, transposed heads vht[((b*H+h)*Dh + d)*S + s]
// mode 2: dst f32 row-major (M x N)
__global__ __launch_bounds__(256) void gemm_bf16_kernel(
    const bf16_t* __restrict__ X, const bf16_t* __restrict__ W,
    const float* __restrict__ bias, void* __restrict__ dst, int mode)
{
  int tile = blockIdx.x * 8 + (threadIdx.x >> 5);  // 8 waves/block, one 16x64 strip each
  int ntn  = HIDDEN / 64;
  int tm   = (tile / ntn) * 16;
  int tn   = (tile % ntn) * 64;
  int lane = threadIdx.x & 31;

  v8f acc[4];
#pragma unroll
  for (int nt = 0; nt < 4; ++nt) {
    float bvv = bias[tn + nt * 16 + (lane & 15)];
#pragma unroll
    for (int i = 0; i < 8; ++i) acc[nt][i] = bvv;
  }

  const bf16_t* arow = X + (size_t)(tm + (lane & 15)) * HIDDEN + ((lane >> 4) << 3);
  for (int k0 = 0; k0 < HIDDEN; k0 += 32) {
    if (k0 + 32 < HIDDEN) __builtin_prefetch(arow + k0 + 32, 0, 3);  // global_prefetch_b8
    v16bf a = frag_from_bf16(X, tm, HIDDEN, k0, lane);
#pragma unroll
    for (int nt = 0; nt < 4; ++nt) {
      v16bf b = frag_from_bf16(W, tn + nt * 16, HIDDEN, k0, lane);  // W rows = W^T cols
      acc[nt] = __builtin_amdgcn_wmma_f32_16x16x32_bf16(false, a, false, b, (short)0,
                                                        acc[nt], false, false);
    }
  }

  // C layout: VGPR i, lanes 0-15 -> (M=i, N=lane); lanes 16-31 -> (M=i+8, N=lane-16)
#pragma unroll
  for (int nt = 0; nt < 4; ++nt) {
    int colg = tn + nt * 16 + (lane & 15);
#pragma unroll
    for (int i = 0; i < 8; ++i) {
      int rowg = tm + i + ((lane >> 4) << 3);
      float v = acc[nt][i];
      if (mode == 2) {
        ((float*)dst)[(size_t)rowg * HIDDEN + colg] = v;
      } else {
        int b_ = rowg / SEQ, s_ = rowg - b_ * SEQ;
        int h_ = colg / HEAD_DIM, d_ = colg - h_ * HEAD_DIM;
        int bh = b_ * HEADS + h_;
        if (mode == 0)
          ((bf16_t*)dst)[(size_t)bh * SEQ * HEAD_DIM + (size_t)s_ * HEAD_DIM + d_] = f2bf(v);
        else
          ((bf16_t*)dst)[((size_t)bh * HEAD_DIM + d_) * SEQ + s_] = f2bf(v);
      }
    }
  }
}

// ---------- attention: one wave per (head, 16-query tile) ----------
__global__ __launch_bounds__(32) void attn_kernel(
    const bf16_t* __restrict__ qh, const bf16_t* __restrict__ kh,
    const bf16_t* __restrict__ vht, const float* __restrict__ shift,
    const int* __restrict__ mask, bf16_t* __restrict__ obf)
{
  __shared__ float Sbuf[16 * SPAD];   // scores, ~66KB
#if HAVE_TDM
  __shared__ float ShLds[16 * SEQ];   // TDM-staged shift tile, 64KB
#endif

  int bh   = blockIdx.x;        // b*HEADS + h (batch-major)
  int qm   = blockIdx.y * 16;   // query tile start
  int lane = threadIdx.x & 31;
  int mrow = bh % BATCH;        // faithful-to-torch mask tiling quirk: (b*H+h) % B
  const float scale = 0.125f;   // 1/sqrt(64)

  const bf16_t* Q  = qh  + (size_t)bh * SEQ * HEAD_DIM;
  const bf16_t* K  = kh  + (size_t)bh * SEQ * HEAD_DIM;
  const bf16_t* Vt = vht + (size_t)bh * HEAD_DIM * SEQ;
  const float*  Sh = shift + ((size_t)bh * SEQ + qm) * SEQ;  // 16 rows of this q-tile

#if HAVE_TDM
  // ---- issue TDM: stage 16 x 1024 fp32 shift tile into LDS, overlapped with QK^T ----
  {
    unsigned lds_off = (unsigned)(unsigned long long)(uintptr_t)(void*)ShLds;  // low 32b = LDS addr
    unsigned long long ga = (unsigned long long)(uintptr_t)Sh;
    u32x4 g0;
    g0[0] = 1u;                                             // count=1 (valid), flags 0
    g0[1] = lds_off;                                        // lds_addr
    g0[2] = (unsigned)(ga & 0xFFFFFFFFu);                   // global_addr[31:0]
    g0[3] = (unsigned)((ga >> 32) & 0x1FFFFFFu) | (2u << 30);  // global_addr[56:32] | type=2
    i32x8 g1;
    g1[0] = (2 << 16);                 // workgroup_mask=0, data_size=2 (4 bytes)
    g1[1] = (int)(1024u << 16);        // tensor_dim0[15:0]=1024 in bits[63:48]
    g1[2] = (int)(16u << 16);          // tensor_dim0[31:16]=0 | tensor_dim1[15:0]=16
    g1[3] = (int)(1024u << 16);        // tensor_dim1[31:16]=0 | tile_dim0=1024
    g1[4] = 16;                        // tile_dim1=16, tile_dim2=0
    g1[5] = 1024;                      // tensor_dim0_stride[31:0]=1024 elements
    g1[6] = 0;                         // stride[47:32]=0, tensor_dim1_stride lo=0
    g1[7] = 0;
    i32x4 z4 = {0, 0, 0, 0};           // VADDR2/3 groups unused (2D tensor)
    i32x8 z8 = {0, 0, 0, 0, 0, 0, 0, 0};
    __builtin_amdgcn_tensor_load_to_lds(g0, g1, z4, z4, z8, 0);  // 6-arg (clang-23) form
  }
#endif

  // ---- pass 1: raw scores = (Q K^T)*scale, masked, stashed in LDS ----
  for (int jt = 0; jt < SEQ / 16; ++jt) {
    v8f acc;
#pragma unroll
    for (int i = 0; i < 8; ++i) acc[i] = 0.0f;
#pragma unroll
    for (int k0 = 0; k0 < HEAD_DIM; k0 += 32) {
      v16bf a = frag_from_bf16(Q, qm,      HEAD_DIM, k0, lane);
      v16bf b = frag_from_bf16(K, jt * 16, HEAD_DIM, k0, lane);  // K rows = K^T cols
      acc = __builtin_amdgcn_wmma_f32_16x16x32_bf16(false, a, false, b, (short)0, acc,
                                                    false, false);
    }
    int col = jt * 16 + (lane & 15);
    int mk  = mask[mrow * SEQ + col];
#pragma unroll
    for (int i = 0; i < 8; ++i) {
      int row = i + ((lane >> 4) << 3);
      int mq  = mask[mrow * SEQ + qm + row];
      float s = acc[i] * scale;
      if ((mq & mk) == 0) s = -1e9f;   // shift added later; -1e9+shift ~ -1e9 for softmax
      Sbuf[row * SPAD + col] = s;
    }
  }

#if HAVE_TDM
  __builtin_amdgcn_s_wait_tensorcnt(0);
#endif
  __syncthreads();

  // ---- fused shift-add + row-max sweep (wave32 reduction) ----
  float my_max = -3.0e38f;  // max of the row this lane owns in A-fragments (row = lane&15)
  for (int r = 0; r < 16; ++r) {
    float m = -3.0e38f;
    for (int c = lane; c < SEQ; c += 32) {
#if HAVE_TDM
      float s = Sbuf[r * SPAD + c] + ShLds[r * SEQ + c];
#else
      float s = Sbuf[r * SPAD + c] + Sh[(size_t)r * SEQ + c];
#endif
      Sbuf[r * SPAD + c] = s;
      m = fmaxf(m, s);
    }
#pragma unroll
    for (int off = 16; off > 0; off >>= 1) m = fmaxf(m, __shfl_xor(m, off, 32));
    if (r == (lane & 15)) my_max = m;
  }
  __syncthreads();

  // ---- pass 2: PV with on-the-fly exp; normalize at the end ----
  v8f oacc[4];
#pragma unroll
  for (int nt = 0; nt < 4; ++nt)
#pragma unroll
    for (int i = 0; i < 8; ++i) oacc[nt][i] = 0.0f;

  float lsum = 0.0f;  // partial sum of exp over this lane's half-row
  for (int k0 = 0; k0 < SEQ; k0 += 32) {
    const float* sp = Sbuf + (lane & 15) * SPAD + k0 + ((lane >> 4) << 3);
    v16bf a;
#pragma unroll
    for (int i = 0; i < 8; ++i) {
      float e0 = __expf(sp[i] - my_max);
      float e1 = __expf(sp[i + 16] - my_max);
      lsum += e0 + e1;
      a[i] = f2bf(e0);
      a[i + 8] = f2bf(e1);
    }
#pragma unroll
    for (int nt = 0; nt < 4; ++nt) {
      v16bf b = frag_from_bf16(Vt, nt * 16, SEQ, k0, lane);  // V^T rows = V cols
      oacc[nt] = __builtin_amdgcn_wmma_f32_16x16x32_bf16(false, a, false, b, (short)0,
                                                         oacc[nt], false, false);
    }
  }
  lsum += __shfl_xor(lsum, 16, 32);  // lanes r and r+16 cover complementary column halves

  // ---- write normalized output in (B,S,HIDDEN) layout, bf16 for the out-proj ----
  int b_ = bh / HEADS, h_ = bh % HEADS;
  int col = lane & 15;
#pragma unroll
  for (int i = 0; i < 8; ++i) {
    int row = i + ((lane >> 4) << 3);
    float denom = __shfl(lsum, row, 32);  // row sums live on lanes 0..15 (dup on 16..31)
    size_t obase = ((size_t)(b_ * SEQ + qm + row)) * HIDDEN + (size_t)h_ * HEAD_DIM;
    float inv = 1.0f / denom;
#pragma unroll
    for (int nt = 0; nt < 4; ++nt)
      obf[obase + nt * 16 + col] = f2bf(oacc[nt][i] * inv);
  }
}

// ---------- host launcher ----------
extern "C" void kernel_launch(void* const* d_in, const int* in_sizes, int n_in,
                              void* d_out, int out_size, void* d_ws, size_t ws_size,
                              hipStream_t stream)
{
  const float* q     = (const float*)d_in[0];
  const float* k     = (const float*)d_in[1];
  const float* v     = (const float*)d_in[2];
  const int*   mask  = (const int*)d_in[3];
  const float* shift = (const float*)d_in[4];
  const float* wq    = (const float*)d_in[5];
  const float* bq    = (const float*)d_in[6];
  const float* wk    = (const float*)d_in[7];
  const float* bk    = (const float*)d_in[8];
  const float* wv    = (const float*)d_in[9];
  const float* bv    = (const float*)d_in[10];
  const float* wo    = (const float*)d_in[11];
  const float* bo    = (const float*)d_in[12];
  float* out = (float*)d_out;

  // workspace layout (bf16): activations 3x2M, weights 4x1M, intermediates 4x2M = 18M elems (36MB)
  size_t nAct = (size_t)M_TOT * HIDDEN;    // 2M
  size_t nW   = (size_t)HIDDEN * HIDDEN;   // 1M
  bf16_t* qbf = (bf16_t*)d_ws;
  bf16_t* kbf = qbf + nAct;
  bf16_t* vbf = kbf + nAct;
  bf16_t* wqb = vbf + nAct;
  bf16_t* wkb = wqb + nW;
  bf16_t* wvb = wkb + nW;
  bf16_t* wob = wvb + nW;
  bf16_t* qh  = wob + nW;
  bf16_t* kh  = qh  + nAct;
  bf16_t* vht = kh  + nAct;
  bf16_t* obf = vht + nAct;

  // one-time fp32 -> bf16 conversion
  cvt_kernel<<<(int)(nAct / 4 / 256), 256, 0, stream>>>(q,  qbf, (int)nAct);
  cvt_kernel<<<(int)(nAct / 4 / 256), 256, 0, stream>>>(k,  kbf, (int)nAct);
  cvt_kernel<<<(int)(nAct / 4 / 256), 256, 0, stream>>>(v,  vbf, (int)nAct);
  cvt_kernel<<<(int)(nW   / 4 / 256), 256, 0, stream>>>(wq, wqb, (int)nW);
  cvt_kernel<<<(int)(nW   / 4 / 256), 256, 0, stream>>>(wk, wkb, (int)nW);
  cvt_kernel<<<(int)(nW   / 4 / 256), 256, 0, stream>>>(wv, wvb, (int)nW);
  cvt_kernel<<<(int)(nW   / 4 / 256), 256, 0, stream>>>(wo, wob, (int)nW);

  int tiles = (M_TOT / 16) * (HIDDEN / 64);  // 2048 16x64 strips
  dim3 gproj(tiles / 8);                     // 8 waves (strips) per 256-thread block

  gemm_bf16_kernel<<<gproj, 256, 0, stream>>>(qbf, wqb, bq, qh,  0);
  gemm_bf16_kernel<<<gproj, 256, 0, stream>>>(kbf, wkb, bk, kh,  0);
  gemm_bf16_kernel<<<gproj, 256, 0, stream>>>(vbf, wvb, bv, vht, 1);

  attn_kernel<<<dim3(BATCH * HEADS, SEQ / 16), 32, 0, stream>>>(qh, kh, vht, shift, mask, obf);

  gemm_bf16_kernel<<<gproj, 256, 0, stream>>>(obf, wob, bo, out, 2);
}
